// Model_79096117723920
// MI455X (gfx1250) — matrix operand order repaired and checked by reading, not compiled, not run on hardware
//
#include <hip/hip_runtime.h>
#include <hip/hip_bf16.h>

// ---------------------------------------------------------------------------
// Types for CDNA5 WMMA (wave32, v_wmma_f32_16x16x32_bf16)
// ---------------------------------------------------------------------------
typedef __bf16 bf16;
typedef bf16  v16bf __attribute__((ext_vector_type(16)));
typedef float v8f   __attribute__((ext_vector_type(8)));

union BF16x16 { uint4 u[2]; v16bf v; };

// ---------------------------------------------------------------------------
// Model dims
// ---------------------------------------------------------------------------
#define BD      64      // batch
#define LEN     512     // encoder seq len (GRU consumes 511)
#define ENC_T   511
#define DEC_L   144
#define PREDL   96
#define DM      1024    // d_model
#define CIN     64      // enc_in == c_out
#define K0C     (CIN + DM)   // 1088  (layer0 fused K)
#define K1C     (DM + DM)    // 2048  (layer1 fused K)

// ---------------------------------------------------------------------------
// Series decomposition: 25-tap moving average with replicate padding
// == mean of x[clamp(t-12..t+12)]
// ---------------------------------------------------------------------------
__global__ void decomp_enc_kernel(const float* __restrict__ x,   // [64,512,64]
                                  bf16* __restrict__ ss_seq,     // [64,511,64]
                                  bf16* __restrict__ st_seq,     // [64,511,64]
                                  bf16* __restrict__ dec_s,      // [64,96,64]
                                  bf16* __restrict__ dec_t) {    // [64,96,64]
    int idx = blockIdx.x * blockDim.x + threadIdx.x;
    if (idx >= BD * LEN * CIN) return;
    int c = idx & 63;
    int t = (idx >> 6) & 511;
    int b = idx >> 15;
    float s = 0.f;
    #pragma unroll
    for (int j = -12; j <= 12; ++j) {
        int tt = t + j; tt = tt < 0 ? 0 : (tt > LEN - 1 ? LEN - 1 : tt);
        s += x[((size_t)b * LEN + tt) * CIN + c];
    }
    float mean = s * (1.0f / 25.0f);
    float seas = x[idx] - mean;
    if (t < ENC_T) {
        ss_seq[((size_t)b * ENC_T + t) * CIN + c] = (bf16)seas;
        st_seq[((size_t)b * ENC_T + t) * CIN + c] = (bf16)mean;
    } else { // t == 511: becomes decoder input step 0
        dec_s[((size_t)b * PREDL + 0) * CIN + c] = (bf16)seas;
        dec_t[((size_t)b * PREDL + 0) * CIN + c] = (bf16)mean;
    }
}

__global__ void decomp_dec_kernel(const float* __restrict__ x,   // [64,144,64]
                                  bf16* __restrict__ dec_s,      // [64,96,64]
                                  bf16* __restrict__ dec_t) {
    int idx = blockIdx.x * blockDim.x + threadIdx.x;
    if (idx >= BD * (PREDL - 1) * CIN) return;
    int c = idx & 63;
    int r = idx >> 6;
    int t = r % (PREDL - 1);
    int b = r / (PREDL - 1);
    float s = 0.f;
    #pragma unroll
    for (int j = -12; j <= 12; ++j) {
        int tt = t + j; tt = tt < 0 ? 0 : (tt > DEC_L - 1 ? DEC_L - 1 : tt);
        s += x[((size_t)b * DEC_L + tt) * CIN + c];
    }
    float mean = s * (1.0f / 25.0f);
    float xv = x[((size_t)b * DEC_L + t) * CIN + c];
    dec_s[((size_t)b * PREDL + 1 + t) * CIN + c] = (bf16)(xv - mean);
    dec_t[((size_t)b * PREDL + 1 + t) * CIN + c] = (bf16)mean;
}

// ---------------------------------------------------------------------------
// Pack GRU weights into fused bf16 matrix Wcat [4096, Kc] (Kc = in + 1024)
//   rows    0..1023 : r  : [Wih_r | Whh_r]
//   rows 1024..2047 : z  : [Wih_z | Whh_z]
//   rows 2048..3071 : nx : [Wih_n |   0  ]
//   rows 3072..4095 : nh : [  0   | Whh_n]
// bcat: r: bih_r+bhh_r, z: bih_z+bhh_z, nx: bih_n, nh: bhh_n
// ---------------------------------------------------------------------------
__global__ void pack_gru_kernel(const float* __restrict__ Wih,
                                const float* __restrict__ Whh,
                                const float* __restrict__ bih,
                                const float* __restrict__ bhh,
                                bf16* __restrict__ Wcat,
                                float* __restrict__ bcat,
                                int in_dim) {
    const int Kc = in_dim + DM;
    const size_t total = (size_t)4096 * Kc;
    for (size_t idx = (size_t)blockIdx.x * blockDim.x + threadIdx.x; idx < total;
         idx += (size_t)gridDim.x * blockDim.x) {
        int j = (int)(idx / Kc);
        int k = (int)(idx % Kc);
        int gate = j >> 10, jj = j & 1023;
        float v = 0.f;
        if (k < in_dim) {
            if (gate != 3) {                      // r,z,nx use Wih
                int srow = ((gate == 2) ? 2048 : gate * 1024) + jj;
                v = Wih[(size_t)srow * in_dim + k];
            }
        } else {
            if (gate != 2) {                      // r,z,nh use Whh
                int srow = ((gate == 3) ? 2048 : gate * 1024) + jj;
                v = Whh[(size_t)srow * DM + (k - in_dim)];
            }
        }
        Wcat[idx] = (bf16)v;
        if (k == 0) {
            float bv;
            if      (gate == 0) bv = bih[jj]        + bhh[jj];
            else if (gate == 1) bv = bih[1024 + jj] + bhh[1024 + jj];
            else if (gate == 2) bv = bih[2048 + jj];
            else                bv = bhh[2048 + jj];
            bcat[j] = bv;
        }
    }
}

// Pack final linear: Wlin [64, 2048] = [Ws | Wt], blin = bs + bt
__global__ void pack_lin_kernel(const float* __restrict__ Ws, const float* __restrict__ bs,
                                const float* __restrict__ Wt, const float* __restrict__ bt,
                                bf16* __restrict__ Wlin, float* __restrict__ blin) {
    int idx = blockIdx.x * blockDim.x + threadIdx.x;
    if (idx >= 64 * 2048) return;
    int n = idx >> 11, k = idx & 2047;
    float v = (k < DM) ? Ws[(size_t)n * DM + k] : Wt[(size_t)n * DM + (k - DM)];
    Wlin[idx] = (bf16)v;
    if (k == 0) blin[n] = bs[n] + bt[n];
}

// ---------------------------------------------------------------------------
// One GRU timestep, both stacks (seasonal=z0, trend=z1) in one launch.
// G = [x_t | h_prev] @ Wcat^T + bcat  -> [64, 4096]
// grid (4,16,2): x = 16-row batch tile, y = 64-col hidden slice, z = stack
// block 256 = 8 waves = 4 gates x 2 K-halves (disjoint K -> no extra global
// traffic, 2 waves/SIMD32 for latency hiding). Partial gate tiles meet in
// LDS; GRU update done locally; h_new -> h_out (+ y_out).
// ---------------------------------------------------------------------------
struct StackArgs {
    const bf16* xseq;   // [B, T, in]  (time-major inner dim = in)
    const bf16* h_in;   // [B, 1024]
    bf16*       h_out;  // [B, 1024]
    bf16*       y_out;  // [B, T, 1024] or nullptr
    const bf16* W;      // [4096, Kc]
    const float* bias;  // [4096]
};

__global__ __launch_bounds__(256) void gru_step_kernel(StackArgs a0, StackArgs a1,
                                                       int t, int T, int in_dim) {
    const StackArgs A = (blockIdx.z == 0) ? a0 : a1;
    const int Kc     = in_dim + DM;
    const int Khalf  = Kc >> 1;                // multiple of 32 (544 / 1024)
    const int mbase  = blockIdx.x * 16;        // batch rows
    const int cg     = blockIdx.y;             // 64-col hidden slice
    const int lane   = threadIdx.x & 31;
    const int l16    = lane & 15;
    const int khalf2 = lane >> 4;              // fragment half (WMMA layout)
    // per-wave-uniform ids -> SGPRs so K loops are scalar (no EXEC juggling)
    const int gate   = __builtin_amdgcn_readfirstlane((threadIdx.x >> 5) & 3);
    const int ksplit = __builtin_amdgcn_readfirstlane(threadIdx.x >> 7);

    __shared__ float Gs[8 * 16 * 64];          // [ksplit][gate] 16x64 f32 partials

    // accumulators; bias only in K-half 0. C-layout: col = lane%16 per lane.
    v8f acc[4];
    #pragma unroll
    for (int nt = 0; nt < 4; ++nt) {
        float bv = (ksplit == 0) ? A.bias[gate * 1024 + cg * 64 + nt * 16 + l16] : 0.f;
        acc[nt] = (v8f){bv, bv, bv, bv, bv, bv, bv, bv};
    }

    const int kbeg = ksplit * Khalf;
    const int kend = kbeg + Khalf;
    // split at x/h boundary (in_dim % 32 == 0): [kbeg,xend) from x, [xend,kend) from h
    const int xend = (kbeg < in_dim) ? ((kend < in_dim) ? kend : in_dim) : kbeg;
    const int nx_it = (xend - kbeg) >> 5;
    const int nh_it = (kend - xend) >> 5;

    // B row pointers, bumped +32 elements (64B) per K-tile
    const bf16* bp0 = A.W + (size_t)(gate * 1024 + cg * 64 + l16) * Kc + kbeg + khalf2 * 16;
    const bf16* bp1 = bp0 + (size_t)16 * Kc;
    const bf16* bp2 = bp0 + (size_t)32 * Kc;
    const bf16* bp3 = bp0 + (size_t)48 * Kc;

    // A fragment pointer (includes per-lane fragment offset)
    const bf16* ap = A.xseq + ((size_t)(mbase + l16) * T + t) * in_dim + kbeg + khalf2 * 16;

    #pragma unroll 1
    for (int rgn = 0; rgn < 2; ++rgn) {
        const int niter = (rgn == 0) ? nx_it : nh_it;
        if (rgn == 1)
            ap = A.h_in + (size_t)(mbase + l16) * DM + (xend - in_dim) + khalf2 * 16;
        for (int it = 0; it < niter; ++it) {
            // load A + all 4 B fragments first -> one big load clause,
            // partial loadcnt waits overlap loads with WMMA issue
            BF16x16 af, b0, b1, b2, b3;
            af.u[0] = *(const uint4*)(ap);
            af.u[1] = *(const uint4*)(ap + 8);
            b0.u[0] = *(const uint4*)(bp0); b0.u[1] = *(const uint4*)(bp0 + 8);
            b1.u[0] = *(const uint4*)(bp1); b1.u[1] = *(const uint4*)(bp1 + 8);
            b2.u[0] = *(const uint4*)(bp2); b2.u[1] = *(const uint4*)(bp2 + 8);
            b3.u[0] = *(const uint4*)(bp3); b3.u[1] = *(const uint4*)(bp3 + 8);
            __builtin_prefetch((const void*)(bp0 + 32), 0, 1);   // next K tiles (L2 hot)
            __builtin_prefetch((const void*)(bp1 + 32), 0, 1);
            __builtin_prefetch((const void*)(bp2 + 32), 0, 1);
            __builtin_prefetch((const void*)(bp3 + 32), 0, 1);
            acc[0] = __builtin_amdgcn_wmma_f32_16x16x32_bf16(
                false, af.v, false, b0.v, (short)0, acc[0], false, false);
            acc[1] = __builtin_amdgcn_wmma_f32_16x16x32_bf16(
                false, af.v, false, b1.v, (short)0, acc[1], false, false);
            acc[2] = __builtin_amdgcn_wmma_f32_16x16x32_bf16(
                false, af.v, false, b2.v, (short)0, acc[2], false, false);
            acc[3] = __builtin_amdgcn_wmma_f32_16x16x32_bf16(
                false, af.v, false, b3.v, (short)0, acc[3], false, false);
            ap += 32; bp0 += 32; bp1 += 32; bp2 += 32; bp3 += 32;
        }
    }

    // ---- spill partial gate tiles to LDS (C layout: m = i + 8*(lane>>4))
    float* Gg = Gs + (ksplit * 4 + gate) * (16 * 64);
    #pragma unroll
    for (int nt = 0; nt < 4; ++nt)
        #pragma unroll
        for (int i = 0; i < 8; ++i)
            Gg[(i + 8 * khalf2) * 64 + nt * 16 + l16] = acc[nt][i];
    __syncthreads();

    // ---- GRU update: r=sig, z=sig, n=tanh(nx + r*nh), h=(1-z)*n + z*h_prev
    for (int e = threadIdx.x; e < 16 * 64; e += 256) {
        int m = e >> 6, j = e & 63;
        float gr  = Gs[0 * 1024 + e] + Gs[4 * 1024 + e];
        float gz  = Gs[1 * 1024 + e] + Gs[5 * 1024 + e];
        float gnx = Gs[2 * 1024 + e] + Gs[6 * 1024 + e];
        float gnh = Gs[3 * 1024 + e] + Gs[7 * 1024 + e];
        float r = 1.f / (1.f + __expf(-gr));
        float z = 1.f / (1.f + __expf(-gz));
        float n = tanhf(gnx + r * gnh);
        int b   = mbase + m;
        int col = cg * 64 + j;
        float hp = (float)A.h_in[(size_t)b * DM + col];
        float hn = (1.f - z) * n + z * hp;
        bf16 hb = (bf16)hn;
        A.h_out[(size_t)b * DM + col] = hb;
        if (A.y_out) A.y_out[((size_t)b * T + t) * DM + col] = hb;
    }
}

// ---------------------------------------------------------------------------
// Final projection: out[6144,64] = [y1_s | y1_t][6144,2048] @ Wlin^T + blin
// grid 96 blocks x 128 threads; wave w -> 16 rows x 64 cols (4 WMMA tiles)
// ---------------------------------------------------------------------------
__global__ __launch_bounds__(128) void final_linear_kernel(
        const bf16* __restrict__ y1s, const bf16* __restrict__ y1t,
        const bf16* __restrict__ Wlin, const float* __restrict__ blin,
        float* __restrict__ out) {
    const int wave  = threadIdx.x >> 5;
    const int lane  = threadIdx.x & 31;
    const int l16   = lane & 15;
    const int khalf = lane >> 4;
    const int mbase = blockIdx.x * 64 + wave * 16;   // row = b*96 + t

    v8f acc[4];
    #pragma unroll
    for (int nt = 0; nt < 4; ++nt) {
        float bv = blin[nt * 16 + l16];
        acc[nt] = (v8f){bv, bv, bv, bv, bv, bv, bv, bv};
    }

    const size_t arow = (size_t)(mbase + l16) * DM;
    const bf16* bq0 = Wlin + (size_t)l16 * 2048 + khalf * 16;
    const bf16* bq1 = bq0 + (size_t)16 * 2048;
    const bf16* bq2 = bq0 + (size_t)32 * 2048;
    const bf16* bq3 = bq0 + (size_t)48 * 2048;

    #pragma unroll 1
    for (int rgn = 0; rgn < 2; ++rgn) {
        const bf16* ap = ((rgn == 0) ? y1s : y1t) + arow + khalf * 16;
        for (int it = 0; it < DM / 32; ++it) {
            BF16x16 af, b0, b1, b2, b3;
            af.u[0] = *(const uint4*)(ap);
            af.u[1] = *(const uint4*)(ap + 8);
            b0.u[0] = *(const uint4*)(bq0); b0.u[1] = *(const uint4*)(bq0 + 8);
            b1.u[0] = *(const uint4*)(bq1); b1.u[1] = *(const uint4*)(bq1 + 8);
            b2.u[0] = *(const uint4*)(bq2); b2.u[1] = *(const uint4*)(bq2 + 8);
            b3.u[0] = *(const uint4*)(bq3); b3.u[1] = *(const uint4*)(bq3 + 8);
            acc[0] = __builtin_amdgcn_wmma_f32_16x16x32_bf16(
                false, af.v, false, b0.v, (short)0, acc[0], false, false);
            acc[1] = __builtin_amdgcn_wmma_f32_16x16x32_bf16(
                false, af.v, false, b1.v, (short)0, acc[1], false, false);
            acc[2] = __builtin_amdgcn_wmma_f32_16x16x32_bf16(
                false, af.v, false, b2.v, (short)0, acc[2], false, false);
            acc[3] = __builtin_amdgcn_wmma_f32_16x16x32_bf16(
                false, af.v, false, b3.v, (short)0, acc[3], false, false);
            ap += 32; bq0 += 32; bq1 += 32; bq2 += 32; bq3 += 32;
        }
    }
    #pragma unroll
    for (int nt = 0; nt < 4; ++nt)
        #pragma unroll
        for (int i = 0; i < 8; ++i)
            out[(size_t)(mbase + i + 8 * khalf) * 64 + nt * 16 + l16] = acc[nt][i];
}

// ---------------------------------------------------------------------------
// Host orchestration
// Input order (flattened from setup_inputs):
//  0 x_enc, 1 x_mark_enc, 2 x_dec, 3 x_mark_dec,
//  4..11  enc_seasonal (L0: Wih,Whh,bih,bhh ; L1: Wih,Whh,bih,bhh)
//  12..19 dec_seasonal, 20 lin_s_W, 21 lin_s_b,
//  22..29 enc_trend, 30..37 dec_trend, 38 lin_t_W, 39 lin_t_b
// ---------------------------------------------------------------------------
extern "C" void kernel_launch(void* const* d_in, const int* in_sizes, int n_in,
                              void* d_out, int out_size, void* d_ws, size_t ws_size,
                              hipStream_t stream) {
    (void)in_sizes; (void)n_in; (void)out_size; (void)ws_size;
    const float* x_enc = (const float*)d_in[0];
    const float* x_dec = (const float*)d_in[2];

    char* ws = (char*)d_ws;
    size_t off = 0;
    auto alloc = [&](size_t bytes) -> char* {
        off = (off + 255) & ~(size_t)255;
        char* p = ws + off; off += bytes; return p;
    };

    // Fused weights: index = set*2 + layer; set: 0=enc_s,1=enc_t,2=dec_s,3=dec_t
    bf16* Wc[8]; float* bc[8];
    for (int i = 0; i < 8; ++i) {
        size_t Kc = (i & 1) ? K1C : K0C;
        Wc[i] = (bf16*)alloc((size_t)4096 * Kc * sizeof(bf16));
        bc[i] = (float*)alloc(4096 * sizeof(float));
    }
    bf16* ss_seq = (bf16*)alloc((size_t)BD * ENC_T * CIN * sizeof(bf16));
    bf16* st_seq = (bf16*)alloc((size_t)BD * ENC_T * CIN * sizeof(bf16));
    bf16* dsq    = (bf16*)alloc((size_t)BD * PREDL * CIN * sizeof(bf16));
    bf16* dtq    = (bf16*)alloc((size_t)BD * PREDL * CIN * sizeof(bf16));
    bf16* y0s    = (bf16*)alloc((size_t)BD * ENC_T * DM * sizeof(bf16));
    bf16* y0t    = (bf16*)alloc((size_t)BD * ENC_T * DM * sizeof(bf16));
    bf16* y1s    = (bf16*)alloc((size_t)BD * PREDL * DM * sizeof(bf16));
    bf16* y1t    = (bf16*)alloc((size_t)BD * PREDL * DM * sizeof(bf16));
    bf16* hb     = (bf16*)alloc((size_t)8 * BD * DM * sizeof(bf16)); // [stack][layer][parity]
    bf16* Wlin   = (bf16*)alloc((size_t)64 * 2048 * sizeof(bf16));
    float* blin  = (float*)alloc(64 * sizeof(float));

    auto H = [&](int s, int l, int p) -> bf16* {
        return hb + (size_t)(((s * 2 + l) * 2) + p) * BD * DM;
    };

    // --- weight packing (runs every call; deterministic) ---
    const int pbase[4] = {4, 22, 12, 30};  // enc_s, enc_t, dec_s, dec_t
    for (int s = 0; s < 4; ++s)
        for (int l = 0; l < 2; ++l) {
            int b0 = pbase[s] + l * 4;
            int in_dim = (l == 0) ? CIN : DM;
            pack_gru_kernel<<<1024, 256, 0, stream>>>(
                (const float*)d_in[b0 + 0], (const float*)d_in[b0 + 1],
                (const float*)d_in[b0 + 2], (const float*)d_in[b0 + 3],
                Wc[s * 2 + l], bc[s * 2 + l], in_dim);
        }
    pack_lin_kernel<<<(64 * 2048 + 255) / 256, 256, 0, stream>>>(
        (const float*)d_in[20], (const float*)d_in[21],
        (const float*)d_in[38], (const float*)d_in[39], Wlin, blin);

    // --- h0 = 0 for all stacks/layers/parities ---
    hipMemsetAsync(hb, 0, (size_t)8 * BD * DM * sizeof(bf16), stream);

    // --- decomposition + decoder input assembly ---
    {
        int tot = BD * LEN * CIN;
        decomp_enc_kernel<<<(tot + 255) / 256, 256, 0, stream>>>(
            x_enc, ss_seq, st_seq, dsq, dtq);
    }
    {
        int tot = BD * (PREDL - 1) * CIN;
        decomp_dec_kernel<<<(tot + 255) / 256, 256, 0, stream>>>(x_dec, dsq, dtq);
    }

    const dim3 sgrid(4, 16, 2);
    // --- encoder layer 0 (seasonal + trend fused) ---
    for (int t = 0; t < ENC_T; ++t) {
        StackArgs a0{ss_seq, H(0,0,t & 1), H(0,0,(t + 1) & 1), y0s, Wc[0], bc[0]};
        StackArgs a1{st_seq, H(1,0,t & 1), H(1,0,(t + 1) & 1), y0t, Wc[2], bc[2]};
        gru_step_kernel<<<sgrid, 256, 0, stream>>>(a0, a1, t, ENC_T, CIN);
    }
    // --- encoder layer 1 (y outputs unused; only final h needed) ---
    for (int t = 0; t < ENC_T; ++t) {
        StackArgs a0{y0s, H(0,1,t & 1), H(0,1,(t + 1) & 1), nullptr, Wc[1], bc[1]};
        StackArgs a1{y0t, H(1,1,t & 1), H(1,1,(t + 1) & 1), nullptr, Wc[3], bc[3]};
        gru_step_kernel<<<sgrid, 256, 0, stream>>>(a0, a1, t, ENC_T, DM);
    }
    // --- decoder layer 0: h0 = encoder final h (parity continues at 511&1) ---
    for (int t = 0; t < PREDL; ++t) {
        int pin = (ENC_T + t) & 1, pout = pin ^ 1;
        StackArgs a0{dsq, H(0,0,pin), H(0,0,pout), y0s, Wc[4], bc[4]};
        StackArgs a1{dtq, H(1,0,pin), H(1,0,pout), y0t, Wc[6], bc[6]};
        gru_step_kernel<<<sgrid, 256, 0, stream>>>(a0, a1, t, PREDL, CIN);
    }
    // --- decoder layer 1 ---
    for (int t = 0; t < PREDL; ++t) {
        int pin = (ENC_T + t) & 1, pout = pin ^ 1;
        StackArgs a0{y0s, H(0,1,pin), H(0,1,pout), y1s, Wc[5], bc[5]};
        StackArgs a1{y0t, H(1,1,pin), H(1,1,pout), y1t, Wc[7], bc[7]};
        gru_step_kernel<<<sgrid, 256, 0, stream>>>(a0, a1, t, PREDL, DM);
    }
    // --- final projection: seasonal + trend in one fused GEMM ---
    final_linear_kernel<<<(BD * PREDL) / 64, 128, 0, stream>>>(
        y1s, y1t, Wlin, blin, (float*)d_out);
}
